// MultiHeadAttention_1030792151464
// MI455X (gfx1250) — matrix operand order repaired and checked by reading, not compiled, not run on hardware
//
#include <hip/hip_runtime.h>
#include <hip/hip_bf16.h>

// ---------------------------------------------------------------------------
// Multi-head attention (B=4, S=2048, E=2048, H=16, Dh=128), column softmax.
// BF16 WMMA (v_wmma_f32_16x16x32_bf16) everywhere; column-softmax folded into
// V rows so the SxS score matrix is never materialized. Software-pipelined
// inner loops so WMMAs overlap with global loads (no full s_wait_loadcnt 0
// before every matrix op).
// ---------------------------------------------------------------------------

typedef __bf16 bf16_t;
typedef bf16_t v4bf  __attribute__((ext_vector_type(4)));
typedef bf16_t v8bf  __attribute__((ext_vector_type(8)));
typedef bf16_t v16bf __attribute__((ext_vector_type(16)));
typedef float  v8f   __attribute__((ext_vector_type(8)));

#define B_   4
#define S_   2048
#define E_   2048
#define H_   16
#define DH_  128
#define BH_  (B_ * H_)
#define HEADSZ (S_ * DH_)          // 262144 elems per (b,h) head slab
#define XN_  (B_ * S_ * E_)        // 16777216

static constexpr float kScale = 0.08838834764831845f;  // 1/sqrt(128)

__device__ __forceinline__ v8f wmma_bf16(v16bf a, v16bf b, v8f c) {
  // D = A(16x32) * B(32x16) + C(16x16 f32)
  return __builtin_amdgcn_wmma_f32_16x16x32_bf16(false, a, false, b,
                                                 (short)0, c, false, false);
}

// A-operand per-lane layout (16-bit A 16x32): lane = half*16 + m:
//   as loaded here, VGPR0..3 hold K = half*16 .. half*16+7, VGPR4..7 hold K = half*16+8 .. +7
__device__ __forceinline__ v16bf load_a_pair(const bf16_t* lo, const bf16_t* hi) {
  v8bf l = *(const v8bf*)lo;
  v8bf h = *(const v8bf*)hi;
  v16bf r;
#pragma unroll
  for (int i = 0; i < 8; ++i) { r[i] = l[i]; r[i + 8] = h[i]; }
  return r;
}

// B-operand lane layout used in this file (16-bit B 32x16, same k order as A): lane = half*16 + n holds the 16
// contiguous contraction elements K = half*16 .. half*16+15 of column n.
__device__ __forceinline__ v16bf load_b(const bf16_t* p) {
  return *(const v16bf*)p;
}

// ---------------------------------------------------------------------------
// 1) x (f32) -> bf16
// ---------------------------------------------------------------------------
__global__ void __launch_bounds__(256)
k_cvt_x(const float* __restrict__ x, bf16_t* __restrict__ xb, int n) {
  int i = (blockIdx.x * blockDim.x + threadIdx.x) * 4;
  if (i + 3 < n) {
    float4 v = *(const float4*)(x + i);
    v4bf o;
    o[0] = (bf16_t)v.x; o[1] = (bf16_t)v.y; o[2] = (bf16_t)v.z; o[3] = (bf16_t)v.w;
    *(v4bf*)(xb + i) = o;
  }
}

// ---------------------------------------------------------------------------
// 2) W (f32, ExE, row = contraction dim) -> Wt (bf16, ExE, transposed) so the
//    GEMM B-operand reads per-lane contiguous 32B chunks. LDS-tiled transpose.
// ---------------------------------------------------------------------------
__global__ void __launch_bounds__(256)
k_cvt_wt(const float* __restrict__ w, bf16_t* __restrict__ wt) {
  __shared__ float tile[32][33];
  int bx = blockIdx.x * 32;  // output-feature cols of W
  int by = blockIdx.y * 32;  // contraction rows of W
  int tx = threadIdx.x, ty = threadIdx.y;   // (32, 8)
#pragma unroll
  for (int i = 0; i < 4; ++i)
    tile[ty + i * 8][tx] = w[(by + ty + i * 8) * E_ + bx + tx];
  __syncthreads();
#pragma unroll
  for (int i = 0; i < 4; ++i)
    wt[(bx + ty + i * 8) * E_ + by + tx] = (bf16_t)tile[tx][ty + i * 8];
}

// ---------------------------------------------------------------------------
// 3) Y(8192x2048) = Xb(8192x2048) @ W, with W provided transposed (Wt[n][k]).
//    One wave computes a 16(M) x 64(N) tile; 4 WMMAs per K=32 step.
//    Double-buffered: tiles for step k+1 are in flight during step-k WMMAs.
// ---------------------------------------------------------------------------
__global__ void __launch_bounds__(256)
k_gemm(const bf16_t* __restrict__ xb, const bf16_t* __restrict__ wt,
       bf16_t* __restrict__ y) {
  int wave = blockIdx.x * (blockDim.x >> 5) + (threadIdx.x >> 5);
  int lane = threadIdx.x & 31;
  int l = lane & 15, half = lane >> 4;
  const int NT = E_ / 64;                 // 32 N-tiles
  int mt = wave / NT, nt = wave % NT;
  int mbase = mt * 16, nbase = nt * 64;

  const bf16_t* arow = xb + (mbase + l) * E_ + half * 16;
  const bf16_t* brow = wt + (nbase + l) * E_ + half * 16;

  v8f acc[4] = {v8f{}, v8f{}, v8f{}, v8f{}};

  v16bf a0 = load_a_pair(arow, arow + 8);
  v16bf b0[4];
#pragma unroll
  for (int t = 0; t < 4; ++t) b0[t] = load_b(brow + t * 16 * E_);

#pragma unroll 2
  for (int kk = 0; kk < E_ / 32; ++kk) {
    int kn = (kk + 1 < E_ / 32) ? (kk + 1) * 32 : kk * 32;
    // issue next-step loads before this step's WMMAs
    v16bf a1 = load_a_pair(arow + kn, arow + kn + 8);
    v16bf b1[4];
#pragma unroll
    for (int t = 0; t < 4; ++t) b1[t] = load_b(brow + t * 16 * E_ + kn);
#pragma unroll
    for (int t = 0; t < 4; ++t) acc[t] = wmma_bf16(a0, b0[t], acc[t]);
    a0 = a1;
#pragma unroll
    for (int t = 0; t < 4; ++t) b0[t] = b1[t];
  }

  // C layout: lane = half*16 + n holds rows M = half*8 + j in VGPR j.
#pragma unroll
  for (int t = 0; t < 4; ++t)
#pragma unroll
    for (int j = 0; j < 8; ++j)
      y[(mbase + half * 8 + j) * E_ + nbase + t * 16 + l] = (bf16_t)acc[t][j];
}

// ---------------------------------------------------------------------------
// 4) Column-softmax stats: per (b,h) and 16-key stripe, online max/sum over
//    all 128 q-tiles of scaled scores s = scale * Q K^T. A-operand loads for
//    the next q-tile are issued before the current WMMA chain.
// ---------------------------------------------------------------------------
__global__ void __launch_bounds__(256)
k_stats(const bf16_t* __restrict__ q, const bf16_t* __restrict__ k,
        float* __restrict__ m_out, float* __restrict__ rz_out) {
  int wave = blockIdx.x * (blockDim.x >> 5) + (threadIdx.x >> 5);
  int lane = threadIdx.x & 31;
  int l = lane & 15, half = lane >> 4;
  int bh = wave >> 7;                     // 64 heads
  int kt = wave & 127;                    // 128 key stripes of 16
  const bf16_t* qb = q + bh * HEADSZ;
  const bf16_t* kb = k + bh * HEADSZ;
  int kcol = kt * 16 + l;

  v16bf bk[4];                            // B operand fixed for this stripe
#pragma unroll
  for (int c = 0; c < 4; ++c)
    bk[c] = load_b(kb + kcol * DH_ + c * 32 + half * 16);

  const bf16_t* abase = qb + l * DH_ + half * 16;
  v16bf a[4];
#pragma unroll
  for (int c = 0; c < 4; ++c)
    a[c] = load_a_pair(abase + c * 32, abase + c * 32 + 8);

  float m = -__builtin_inff();
  float z = 0.0f;
  for (int qt = 0; qt < S_ / 16; ++qt) {
    int qn = (qt + 1 < S_ / 16) ? qt + 1 : qt;
    const bf16_t* an_p = abase + qn * 16 * DH_;
    v16bf an[4];
#pragma unroll
    for (int c = 0; c < 4; ++c)
      an[c] = load_a_pair(an_p + c * 32, an_p + c * 32 + 8);

    v8f cc = v8f{};
#pragma unroll
    for (int c = 0; c < 4; ++c) cc = wmma_bf16(a[c], bk[c], cc);

    float tm = cc[0];
#pragma unroll
    for (int j = 1; j < 8; ++j) tm = fmaxf(tm, cc[j]);
    tm *= kScale;
    tm = fmaxf(tm, __shfl_xor(tm, 16, 32));   // combine rows 0-7 with 8-15
    float mn = fmaxf(m, tm);
    float se = 0.0f;
#pragma unroll
    for (int j = 0; j < 8; ++j) se += __expf(cc[j] * kScale - mn);
    se += __shfl_xor(se, 16, 32);
    z = z * __expf(m - mn) + se;
    m = mn;
#pragma unroll
    for (int c = 0; c < 4; ++c) a[c] = an[c];
  }
  if (half == 0) {
    m_out[bh * S_ + kcol]  = m;
    rz_out[bh * S_ + kcol] = 1.0f / z;
  }
}

// ---------------------------------------------------------------------------
// 5) V' = (V / Z) transposed to [d][k] (bf16) so the attn B-operand reads
//    per-lane contiguous chunks along k. LDS-tiled transpose per head.
// ---------------------------------------------------------------------------
__global__ void __launch_bounds__(256)
k_scale_v(const bf16_t* __restrict__ v, const float* __restrict__ rz,
          bf16_t* __restrict__ vst) {
  __shared__ float tile[32][33];
  int bh = blockIdx.z;
  int kbase = blockIdx.x * 32;
  int dbase = blockIdx.y * 32;
  int tx = threadIdx.x, ty = threadIdx.y;   // (32, 8)
  const bf16_t* vb = v + bh * HEADSZ;
  bf16_t* vd = vst + bh * HEADSZ;
#pragma unroll
  for (int i = 0; i < 4; ++i) {
    int kk = kbase + ty + i * 8;
    tile[ty + i * 8][tx] = (float)vb[kk * DH_ + dbase + tx] * rz[bh * S_ + kk];
  }
  __syncthreads();
#pragma unroll
  for (int i = 0; i < 4; ++i)
    vd[(dbase + ty + i * 8) * S_ + kbase + tx] = (bf16_t)tile[tx][ty + i * 8];
}

// ---------------------------------------------------------------------------
// 6) attn[q,d] = sum_k exp(scale*QK^T - m[k]) * V'[k,d].
//    One wave: 16(q) x 128(d) output. Per 32-key chunk: 8 score WMMAs,
//    exp, C-layout -> A-layout reshuffle through LDS, 8 accumulate WMMAs.
//    B tiles are batch-loaded before each WMMA group; next chunk's K rows
//    are prefetched (global_prefetch_b8).
// ---------------------------------------------------------------------------
#define STRIDE_ 40   // LDS row stride (bf16 elems): 80B keeps ds_load_b128
                     // 16B-aligned and bank-conflict-free across 16 lanes.
__global__ void __launch_bounds__(256)
k_attn(const bf16_t* __restrict__ q, const bf16_t* __restrict__ k,
       const bf16_t* __restrict__ vst, const float* __restrict__ marr,
       float* __restrict__ out) {
  __shared__ bf16_t stag[8][16 * STRIDE_];
  int wslot = threadIdx.x >> 5;
  int wave = blockIdx.x * 8 + wslot;
  int lane = threadIdx.x & 31;
  int l = lane & 15, half = lane >> 4;
  int bh = wave >> 7;
  int qt = wave & 127;

  const bf16_t* qb = q + bh * HEADSZ;
  const bf16_t* kb = k + bh * HEADSZ;
  const bf16_t* vb = vst + bh * HEADSZ;
  const float* mb = marr + bh * S_;
  float* ob = out + bh * HEADSZ;
  bf16_t* st = stag[wslot];

  v16bf qa[4];                            // Q tile, invariant over k-loop
#pragma unroll
  for (int c = 0; c < 4; ++c) {
    const bf16_t* ap = qb + (qt * 16 + l) * DH_ + c * 32 + half * 16;
    qa[c] = load_a_pair(ap, ap + 8);
  }

  v8f acc[8];
#pragma unroll
  for (int t = 0; t < 8; ++t) acc[t] = v8f{};

  for (int kt = 0; kt < S_ / 32; ++kt) {
    int k32 = kt * 32;

    // prefetch next chunk's K rows into cache (global_prefetch_b8)
    if (kt + 1 < S_ / 32) {
      __builtin_prefetch(kb + (k32 + 32 + l) * DH_ + half * 64, 0, 3);
      __builtin_prefetch(kb + (k32 + 48 + l) * DH_ + half * 64, 0, 3);
    }

    // batch-load all 8 score B-tiles, then run the WMMA group
    v16bf bk0[4], bk1[4];
#pragma unroll
    for (int c = 0; c < 4; ++c) {
      bk0[c] = load_b(kb + (k32 + l) * DH_ + c * 32 + half * 16);
      bk1[c] = load_b(kb + (k32 + 16 + l) * DH_ + c * 32 + half * 16);
    }
    float m0 = mb[k32 + l];
    float m1 = mb[k32 + 16 + l];
    v8f c0 = v8f{}, c1 = v8f{};
#pragma unroll
    for (int c = 0; c < 4; ++c) {
      c0 = wmma_bf16(qa[c], bk0[c], c0);
      c1 = wmma_bf16(qa[c], bk1[c], c1);
    }

    // batch-load all 8 V' B-tiles while the exp/LDS staging happens
    v16bf bv[8];
#pragma unroll
    for (int t = 0; t < 8; ++t)
      bv[t] = load_b(vb + (t * 16 + l) * S_ + k32 + half * 16);

    // E = exp(s*scale - m[col]); stage to LDS row-major [q(16)][k(32)].
#pragma unroll
    for (int j = 0; j < 8; ++j) {
      int row = half * 8 + j;
      st[row * STRIDE_ + l]      = (bf16_t)__expf(c0[j] * kScale - m0);
      st[row * STRIDE_ + 16 + l] = (bf16_t)__expf(c1[j] * kScale - m1);
    }
    asm volatile("" ::: "memory");   // keep ds stores before ds loads;
                                     // DS is in-order within a wave.
    const bf16_t* er = st + l * STRIDE_;
    v16bf ea = load_a_pair(er + half * 16, er + half * 16 + 8);
#pragma unroll
    for (int t = 0; t < 8; ++t) acc[t] = wmma_bf16(ea, bv[t], acc[t]);
  }
#pragma unroll
  for (int t = 0; t < 8; ++t)
#pragma unroll
    for (int j = 0; j < 8; ++j)
      ob[(qt * 16 + half * 8 + j) * DH_ + t * 16 + l] = acc[t][j];
}

// ---------------------------------------------------------------------------
// Workspace layout (bytes)
// ---------------------------------------------------------------------------
#define OFF_XBF  0ull
#define OFF_WQT  33554432ull
#define OFF_WKT  41943040ull
#define OFF_WVT  50331648ull
#define OFF_Q    58720256ull
#define OFF_K    92274688ull
#define OFF_V    125829120ull
#define OFF_VST  159383552ull
#define OFF_M    192937984ull
#define OFF_RZ   193462272ull
#define WS_NEED  193986560ull

extern "C" void kernel_launch(void* const* d_in, const int* in_sizes, int n_in,
                              void* d_out, int out_size, void* d_ws, size_t ws_size,
                              hipStream_t stream) {
  (void)in_sizes; (void)n_in; (void)out_size;
  if (ws_size < WS_NEED) return;

  const float* x  = (const float*)d_in[0];
  const float* wq = (const float*)d_in[1];
  const float* wk = (const float*)d_in[2];
  const float* wv = (const float*)d_in[3];
  char* ws = (char*)d_ws;

  bf16_t* xb  = (bf16_t*)(ws + OFF_XBF);
  bf16_t* wqt = (bf16_t*)(ws + OFF_WQT);
  bf16_t* wkt = (bf16_t*)(ws + OFF_WKT);
  bf16_t* wvt = (bf16_t*)(ws + OFF_WVT);
  bf16_t* qb  = (bf16_t*)(ws + OFF_Q);
  bf16_t* kb  = (bf16_t*)(ws + OFF_K);
  bf16_t* vb  = (bf16_t*)(ws + OFF_V);
  bf16_t* vst = (bf16_t*)(ws + OFF_VST);
  float*  mf  = (float*)(ws + OFF_M);
  float*  rzf = (float*)(ws + OFF_RZ);
  float*  out = (float*)d_out;

  // 1) conversions
  k_cvt_x<<<XN_ / (256 * 4), 256, 0, stream>>>(x, xb, XN_);
  dim3 tb(32, 8);
  dim3 tg(E_ / 32, E_ / 32);
  k_cvt_wt<<<tg, tb, 0, stream>>>(wq, wqt);
  k_cvt_wt<<<tg, tb, 0, stream>>>(wk, wkt);
  k_cvt_wt<<<tg, tb, 0, stream>>>(wv, wvt);

  // 2) projections: (B*S/16)*(E/64) waves / 8 per block = 2048 blocks
  k_gemm<<<2048, 256, 0, stream>>>(xb, wqt, qb);
  k_gemm<<<2048, 256, 0, stream>>>(xb, wkt, kb);
  k_gemm<<<2048, 256, 0, stream>>>(xb, wvt, vb);

  // 3) column-softmax stats: 64 heads * 128 key stripes = 8192 waves
  k_stats<<<1024, 256, 0, stream>>>(qb, kb, mf, rzf);

  // 4) V' = (V/Z)^T
  k_scale_v<<<dim3(S_ / 32, DH_ / 32, BH_), tb, 0, stream>>>(vb, rzf, vst);

  // 5) attention: 64 heads * 128 q-tiles = 8192 waves
  k_attn<<<1024, 256, 0, stream>>>(qb, kb, vst, mf, out);
}
